// SparseAttention_30812095381509
// MI455X (gfx1250) — compile-verified, matrix operand order checked by reading, not run on hardware
//
#include <hip/hip_runtime.h>
#include <hip/hip_bf16.h>
#include <stdint.h>

typedef __attribute__((ext_vector_type(16))) __bf16 v16bf;
typedef __attribute__((ext_vector_type(8)))  float  v8f;

union FragA {
    v16bf v;
    uint4 q[2];
    unsigned int u[8];
    unsigned short h[16];
};

__device__ __forceinline__ unsigned short f32_bf16(float f) {
    unsigned int u = __float_as_uint(f);
    u += 0x7FFFu + ((u >> 16) & 1u);   // round-to-nearest-even
    return (unsigned short)(u >> 16);
}

__device__ __forceinline__ v8f wmma_bf16(const FragA& a, const FragA& b, v8f c) {
    return __builtin_amdgcn_wmma_f32_16x16x32_bf16(
        false, a.v, false, b.v, (short)0, c, false, false);
}

// ---------------- conversion kernels ----------------
__global__ void cvt_bf16(const float* __restrict__ in, unsigned short* __restrict__ out, int n) {
    int i = blockIdx.x * blockDim.x + threadIdx.x;
    if (i < n) out[i] = f32_bf16(in[i]);
}

// W [K][N] row-major fp32  ->  Wt [N][K] bf16
__global__ void cvt_transpose_bf16(const float* __restrict__ W, unsigned short* __restrict__ Wt,
                                   int K, int N) {
    int i = blockIdx.x * blockDim.x + threadIdx.x;
    if (i < K * N) {
        int k = i / N, n = i - k * N;
        Wt[(size_t)n * K + k] = f32_bf16(W[i]);
    }
}

// ---------------- GEMM: C[M,N] = A[M,K] * Wt[N,K]^T + bias ----------------
// A, Wt bf16 (ushort). 256 threads = 8 waves. Tile 128x128, K-step 32.
template <bool OUT_BF16>
__global__ __launch_bounds__(256, 2) void gemm_bf16(
        const unsigned short* __restrict__ A,
        const unsigned short* __restrict__ Wt,
        const float* __restrict__ bias,
        void* __restrict__ Cout,
        int M, int N, int K) {
    __shared__ __align__(16) unsigned short ldsA[128 * 32];
    __shared__ __align__(16) unsigned short ldsB[128 * 32];

    const int tid  = threadIdx.x;
    const int lane = tid & 31;
    const int wave = tid >> 5;
    const int n0 = blockIdx.x * 128;
    const int m0 = blockIdx.y * 128;

    const int kb    = (lane >= 16) ? 8 : 0;   // K-base within A/B fragment halves
    const int half8 = (lane >= 16) ? 8 : 0;   // C-matrix row offset

    v8f acc[8];
#pragma unroll
    for (int j = 0; j < 8; ++j)
#pragma unroll
        for (int i = 0; i < 8; ++i) acc[j][i] = 0.f;

    const int rowA = wave * 16 + (lane & 15);

    for (int kk = 0; kk < K; kk += 32) {
        // A tile: 128 rows x 32 k = 512 uint4
#pragma unroll
        for (int p = 0; p < 2; ++p) {
            int e = tid + p * 256;
            int r = e >> 2, seg = e & 3;
            const unsigned short* ga = A + (size_t)(m0 + r) * K + kk + seg * 8;
            ((uint4*)ldsA)[e] = *(const uint4*)ga;
            if (kk + 32 < K) __builtin_prefetch(ga + 32, 0, 0);  // global_prefetch_b8
        }
        // B tile: 128 cols x 32 k = 512 uint4
#pragma unroll
        for (int p = 0; p < 2; ++p) {
            int e = tid + p * 256;
            int r = e >> 2, seg = e & 3;
            const unsigned short* gw = Wt + (size_t)(n0 + r) * K + kk + seg * 8;
            ((uint4*)ldsB)[e] = *(const uint4*)gw;
            if (kk + 32 < K) __builtin_prefetch(gw + 32, 0, 0);
        }
        __syncthreads();

        FragA fa;
        fa.q[0] = *(const uint4*)(ldsA + rowA * 32 + kb);
        fa.q[1] = *(const uint4*)(ldsA + rowA * 32 + kb + 16);

#pragma unroll
        for (int j = 0; j < 8; ++j) {
            int col = j * 16 + (lane & 15);
            FragA fb;
            fb.q[0] = *(const uint4*)(ldsB + col * 32 + kb);
            fb.q[1] = *(const uint4*)(ldsB + col * 32 + kb + 16);
            acc[j] = wmma_bf16(fa, fb, acc[j]);
        }
        __syncthreads();
    }

#pragma unroll
    for (int j = 0; j < 8; ++j) {
        int col = n0 + j * 16 + (lane & 15);
        float bv = bias[col];
#pragma unroll
        for (int i = 0; i < 8; ++i) {
            int row = m0 + wave * 16 + half8 + i;
            float v = acc[j][i] + bv;
            if (OUT_BF16)
                ((unsigned short*)Cout)[(size_t)row * N + col] = f32_bf16(v);
            else
                ((float*)Cout)[(size_t)row * N + col] = v;
        }
    }
}

// ---------------- windowed attention (flash-style, 2 chunks of 128 cols) ----
// qkv bf16 [32768][1536]; out bf16 [32768][512]
// grid = b*h*nw = 1024 blocks, 512 threads = 16 waves, 16 q-rows per wave.
#define VTP 136   // V^T row pitch (pad vs bank conflicts; 272B = 16B-aligned rows)

__global__ __launch_bounds__(512, 1) void attn_win(
        const unsigned short* __restrict__ qkv,
        const float* __restrict__ attn_bias,
        unsigned short* __restrict__ out) {
    __shared__ __align__(16) unsigned short ldsK[128 * 64];   // K chunk [128 rows][64 d]
    __shared__ __align__(16) unsigned short ldsVT[64 * VTP];  // V^T chunk [64 d][128 rows]
    __shared__ __align__(16) unsigned short ldsP[16 * 512];   // per-wave 16x32 P scratch

    const int tid  = threadIdx.x;
    const int lane = tid & 31;
    const int wave = tid >> 5;
    const int bid  = blockIdx.x;
    const int wi = bid & 31;
    const int hh = (bid >> 5) & 7;
    const int bb = bid >> 8;
    const int bn0 = bb * 8192 + wi * 256;

    const int kb    = (lane >= 16) ? 8 : 0;
    const int half8 = (lane >= 16) ? 8 : 0;
    const int r0    = wave * 16;

    // Q fragments (A layout): rows r0..r0+15, d = 0..63 in two K=32 chunks
    FragA fq[2];
    {
        const unsigned short* g = qkv + (size_t)(bn0 + r0 + (lane & 15)) * 1536 + hh * 64;
#pragma unroll
        for (int s = 0; s < 2; ++s) {
            fq[s].q[0] = *(const uint4*)(g + s * 32 + kb);
            fq[s].q[1] = *(const uint4*)(g + s * 32 + kb + 16);
        }
    }

    const float scale = 0.125f;   // 64^-0.5
    const float* gb = attn_bias + hh * 65536;

    float m[8], sum[8];
#pragma unroll
    for (int i = 0; i < 8; ++i) { m[i] = -3.0e38f; sum[i] = 0.f; }

    v8f O[4];
#pragma unroll
    for (int j = 0; j < 4; ++j)
#pragma unroll
        for (int i = 0; i < 8; ++i) O[j][i] = 0.f;

    unsigned short* ps = ldsP + wave * 512;

    for (int cc = 0; cc < 2; ++cc) {
        const int kc0 = cc * 128;   // chunk column base
        __syncthreads();            // previous chunk fully consumed

        // load K chunk [128][64]: 1024 uint4
        {
            const unsigned short* gk = qkv + (size_t)(bn0 + kc0) * 1536 + 512 + hh * 64;
#pragma unroll
            for (int p = 0; p < 2; ++p) {
                int e = tid + p * 512;
                int kr = e >> 3, seg = e & 7;
                const unsigned short* src = gk + (size_t)kr * 1536 + seg * 8;
                ((uint4*)ldsK)[e] = *(const uint4*)src;
                if (cc == 0) __builtin_prefetch(src + 128 * 1536, 0, 0);  // next chunk
            }
        }
        // load V^T chunk [64][128] (pitch VTP): 8192 elems, coalesced reads
        {
            const unsigned short* gv = qkv + (size_t)(bn0 + kc0) * 1536 + 1024 + hh * 64;
#pragma unroll
            for (int p = 0; p < 16; ++p) {
                int e = tid + p * 512;
                int kr = e >> 6, d = e & 63;
                ldsVT[d * VTP + kr] = gv[(size_t)kr * 1536 + d];
            }
            if (cc == 0) __builtin_prefetch(gv + (size_t)(128 + (tid >> 2)) * 1536, 0, 0);
        }
        __syncthreads();

        // S chunk = Q K^T : 8 tiles of 16x16
        v8f S[8];
#pragma unroll
        for (int t = 0; t < 8; ++t)
#pragma unroll
            for (int i = 0; i < 8; ++i) S[t][i] = 0.f;

#pragma unroll
        for (int t = 0; t < 8; ++t) {
            const unsigned short* kp = ldsK + (t * 16 + (lane & 15)) * 64;
            FragA fb0, fb1;
            fb0.q[0] = *(const uint4*)(kp + kb);
            fb0.q[1] = *(const uint4*)(kp + kb + 16);
            fb1.q[0] = *(const uint4*)(kp + 32 + kb);
            fb1.q[1] = *(const uint4*)(kp + 32 + kb + 16);
            S[t] = wmma_bf16(fq[0], fb0, S[t]);
            S[t] = wmma_bf16(fq[1], fb1, S[t]);
        }

        // scale + bias
#pragma unroll
        for (int t = 0; t < 8; ++t) {
            int kcol = kc0 + t * 16 + (lane & 15);
#pragma unroll
            for (int i = 0; i < 8; ++i) {
                int qrow = r0 + half8 + i;
                S[t][i] = S[t][i] * scale + gb[qrow * 256 + kcol];
            }
        }

        // online softmax update
        float cm[8];
#pragma unroll
        for (int i = 0; i < 8; ++i) cm[i] = m[i];
#pragma unroll
        for (int t = 0; t < 8; ++t)
#pragma unroll
            for (int i = 0; i < 8; ++i) cm[i] = fmaxf(cm[i], S[t][i]);
#pragma unroll
        for (int msk = 1; msk < 16; msk <<= 1)
#pragma unroll
            for (int i = 0; i < 8; ++i) cm[i] = fmaxf(cm[i], __shfl_xor(cm[i], msk, 32));

        float f[8];
#pragma unroll
        for (int i = 0; i < 8; ++i) {
            f[i] = __expf(m[i] - cm[i]);   // 0 on first chunk
            m[i] = cm[i];
            sum[i] *= f[i];
        }
#pragma unroll
        for (int j = 0; j < 4; ++j)
#pragma unroll
            for (int i = 0; i < 8; ++i) O[j][i] *= f[i];

        float cs[8];
#pragma unroll
        for (int i = 0; i < 8; ++i) cs[i] = 0.f;
#pragma unroll
        for (int t = 0; t < 8; ++t)
#pragma unroll
            for (int i = 0; i < 8; ++i) {
                S[t][i] = __expf(S[t][i] - m[i]);
                cs[i] += S[t][i];
            }
#pragma unroll
        for (int msk = 1; msk < 16; msk <<= 1)
#pragma unroll
            for (int i = 0; i < 8; ++i) cs[i] += __shfl_xor(cs[i], msk, 32);
#pragma unroll
        for (int i = 0; i < 8; ++i) sum[i] += cs[i];

        // O += P V over this chunk: 4 sub-chunks of K=32
#pragma unroll
        for (int c4 = 0; c4 < 4; ++c4) {
#pragma unroll
            for (int tt = 0; tt < 2; ++tt) {
                int t = 2 * c4 + tt;
                int col = tt * 16 + (lane & 15);
#pragma unroll
                for (int i = 0; i < 8; ++i)
                    ps[(half8 + i) * 32 + col] = f32_bf16(S[t][i]);
            }
            FragA fp;
            fp.q[0] = *(const uint4*)(ps + (lane & 15) * 32 + kb);
            fp.q[1] = *(const uint4*)(ps + (lane & 15) * 32 + kb + 16);
#pragma unroll
            for (int j = 0; j < 4; ++j) {
                const unsigned short* vp = ldsVT + (j * 16 + (lane & 15)) * VTP + c4 * 32;
                FragA fv;
                fv.q[0] = *(const uint4*)(vp + kb);
                fv.q[1] = *(const uint4*)(vp + kb + 16);
                O[j] = wmma_bf16(fp, fv, O[j]);
            }
        }
    }

    // final normalize + store bf16 into [32768][512] at columns hh*64..+63
    float inv[8];
#pragma unroll
    for (int i = 0; i < 8; ++i) inv[i] = 1.f / sum[i];
#pragma unroll
    for (int j = 0; j < 4; ++j) {
        int col = hh * 64 + j * 16 + (lane & 15);
#pragma unroll
        for (int i = 0; i < 8; ++i) {
            int row = bn0 + r0 + half8 + i;
            out[(size_t)row * 512 + col] = f32_bf16(O[j][i] * inv[i]);
        }
    }
}

// ---------------- launch ----------------
extern "C" void kernel_launch(void* const* d_in, const int* in_sizes, int n_in,
                              void* d_out, int out_size, void* d_ws, size_t ws_size,
                              hipStream_t stream) {
    const float* x         = (const float*)d_in[0];
    const float* Wqkv      = (const float*)d_in[1];
    const float* bqkv      = (const float*)d_in[2];
    const float* Wproj     = (const float*)d_in[3];
    const float* bproj     = (const float*)d_in[4];
    const float* attn_bias = (const float*)d_in[5];
    float* out = (float*)d_out;

    const int M = 4 * 8192;  // 32768 rows

    // workspace layout (bf16/ushort)
    unsigned short* x_bf   = (unsigned short*)d_ws;
    unsigned short* WqkvT  = x_bf + (size_t)M * 512;
    unsigned short* WprojT = WqkvT + (size_t)1536 * 512;
    unsigned short* qkv    = WprojT + (size_t)512 * 512;
    unsigned short* att    = qkv + (size_t)M * 1536;

    cvt_bf16<<<(M * 512 + 255) / 256, 256, 0, stream>>>(x, x_bf, M * 512);
    cvt_transpose_bf16<<<(512 * 1536 + 255) / 256, 256, 0, stream>>>(Wqkv, WqkvT, 512, 1536);
    cvt_transpose_bf16<<<(512 * 512 + 255) / 256, 256, 0, stream>>>(Wproj, WprojT, 512, 512);

    // qkv = x @ Wqkv + bqkv  (bf16 out)
    gemm_bf16<true><<<dim3(1536 / 128, M / 128), 256, 0, stream>>>(
        x_bf, WqkvT, bqkv, qkv, M, 1536, 512);

    // windowed attention -> att (bf16)
    attn_win<<<4 * 8 * 32, 512, 0, stream>>>(qkv, attn_bias, att);

    // out = att @ Wproj + bproj (fp32 out)
    gemm_bf16<false><<<dim3(512 / 128, M / 128), 256, 0, stream>>>(
        att, WprojT, bproj, out, M, 512, 512);
}
